// _TransformerBlockQuantum_65481071402755
// MI455X (gfx1250) — matrix operand order repaired and checked by reading, not compiled, not run on hardware
//
#include <hip/hip_runtime.h>
#include <hip/hip_bf16.h>
#include <hip/hip_fp16.h>

// ---- shapes ----
#define Bz 8
#define Tz 2048
#define Ez 64
#define Hz 8
#define DKz 8
#define FFz 256
#define BT (Bz * Tz)

typedef __attribute__((ext_vector_type(16))) _Float16 v16h;
typedef __attribute__((ext_vector_type(8)))  float    v8f;

#if __has_builtin(__builtin_amdgcn_global_load_async_to_lds_b128)
#define HAVE_ASYNC_LDS 1
typedef int v4i __attribute__((vector_size(16)));
typedef __attribute__((address_space(1))) v4i glb_v4i;
typedef __attribute__((address_space(3))) v4i lds_v4i;
// async 16-byte copy global -> LDS (per active lane), tracked by ASYNCcnt
__device__ __forceinline__ void async_copy16(const _Float16* gsrc, _Float16* ldst) {
  __builtin_amdgcn_global_load_async_to_lds_b128(
      (glb_v4i*)(gsrc), (lds_v4i*)(ldst), 0, 0);
}
#endif

// D = A(16x32 f16) * B(32x16 f16) + C(16x16 f32)
__device__ __forceinline__ v8f wmma_f16(v16h a, v16h b, v8f c) {
  return __builtin_amdgcn_wmma_f32_16x16x32_f16(
      false, a, false, b, (short)0, c, false, false);
}

// A-matrix fragment gather from row-major f16 LDS tile.
// Layout (contiguous K, matching B loaders): lanes 0-15: M=lane, K=0..15;
// lanes 16-31: M=lane-16, K=16..31.
__device__ __forceinline__ v16h load_A(const _Float16* lds, int ld, int kbase) {
  int lane = threadIdx.x & 31;
  int m = lane & 15;
  int khalf = (lane >> 4) << 4;                 // 0 or 16
  const _Float16* p = lds + m * ld + kbase + khalf;
  v16h a;
#pragma unroll
  for (int i = 0; i < 8; ++i) a[i] = p[i];
#pragma unroll
  for (int i = 0; i < 8; ++i) a[8 + i] = p[8 + i];
  return a;
}

// B-matrix fragment, B[k][n] = Wrows[nbase+n][kbase+k] (row-major, k contiguous).
__device__ __forceinline__ v16h load_B_row(const _Float16* lds, int ld, int nbase, int kbase) {
  int lane = threadIdx.x & 31;
  int n = lane & 15;
  int kset = (lane >> 4) << 4;                  // 0 or 16
  const _Float16* p = lds + (nbase + n) * ld + kbase + kset;
  v16h b;
#pragma unroll
  for (int i = 0; i < 16; ++i) b[i] = p[i];
  return b;
}

// B-matrix fragment from a [32][ld] tile with K as the row index: B[k][n] = tile[k][n].
__device__ __forceinline__ v16h load_B_col(const _Float16* lds, int ld) {
  int lane = threadIdx.x & 31;
  int n = lane & 15;
  int kset = (lane >> 4) << 4;
  v16h b;
#pragma unroll
  for (int i = 0; i < 16; ++i) b[i] = lds[(kset + i) * ld + n];
  return b;
}

// ---------------------------------------------------------------------------
// Kernel 1: q/k/v = cos(x @ W^T + b + theta)  -> f16 [BT, 64]
// block = 256 (8 waves), 64 rows per block; 48 tile-jobs (3 mats x 4 rt x 4 nt)
// ---------------------------------------------------------------------------
__global__ void __launch_bounds__(256)
qkv_kernel(const float* __restrict__ x,
           const float* __restrict__ Wq, const float* __restrict__ bq,
           const float* __restrict__ Wk, const float* __restrict__ bk,
           const float* __restrict__ Wv, const float* __restrict__ bv,
           const float* __restrict__ theta_attn,
           _Float16* __restrict__ qg, _Float16* __restrict__ kg,
           _Float16* __restrict__ vg) {
  __shared__ _Float16 sW[3][Ez * Ez];   // 24 KB
  __shared__ _Float16 sX[64 * Ez];      // 8 KB
  __shared__ float    sB[3 * Ez];
  const int tid = threadIdx.x;
  const int row0 = blockIdx.x * 64;

  const float* Ws[3] = {Wq, Wk, Wv};
#pragma unroll
  for (int mi = 0; mi < 3; ++mi)
    for (int i = tid; i < Ez * Ez; i += 256) sW[mi][i] = (_Float16)Ws[mi][i];
  for (int i = tid; i < 64 * Ez; i += 256) sX[i] = (_Float16)x[(size_t)row0 * Ez + i];
  for (int i = tid; i < 3 * Ez; i += 256)
    sB[i] = (i < 64) ? bq[i] : ((i < 128) ? bk[i - 64] : bv[i - 128]);
  __syncthreads();

  const float th = *theta_attn;
  const int wave = tid >> 5, lane = tid & 31;
  const int half = lane >> 4, n = lane & 15;

  for (int job = wave; job < 48; job += 8) {
    const int mat = job >> 4, rt = (job >> 2) & 3, nbase = (job & 3) * 16;
    v8f acc = {};
#pragma unroll
    for (int kc = 0; kc < 2; ++kc) {
      v16h a = load_A(sX + rt * 16 * Ez, Ez, kc * 32);
      v16h b = load_B_row(sW[mat], Ez, nbase, kc * 32);
      acc = wmma_f16(a, b, acc);
    }
    const float bias = sB[mat * Ez + nbase + n];
    _Float16* dst = (mat == 0) ? qg : ((mat == 1) ? kg : vg);
#pragma unroll
    for (int r = 0; r < 8; ++r) {
      const int m = rt * 16 + r + half * 8;
      dst[(size_t)(row0 + m) * Ez + nbase + n] = (_Float16)__cosf(acc[r] + bias + th);
    }
  }
}

// ---------------------------------------------------------------------------
// Kernel 2: flash attention per (b,h). block = 256 (8 waves) covers 128 Q rows
// (16 rows/wave). Shared, double-buffered K/V tiles (one producer wave) loaded
// via GLOBAL_LOAD_ASYNC_TO_LDS_B128 when available; online softmax; DK=8
// zero-padded to WMMA K=32.
// ---------------------------------------------------------------------------
__global__ void __launch_bounds__(256)
attn_kernel(const _Float16* __restrict__ qg, const _Float16* __restrict__ kg,
            const _Float16* __restrict__ vg, _Float16* __restrict__ og) {
  __shared__ _Float16 sQ[8][16 * 32];   // per-wave Q tile (padded K)
  __shared__ _Float16 sP[8][16 * 32];   // per-wave P staging (padded K)
  __shared__ _Float16 sK[2][16 * 32];   // shared, double-buffered (padded K)
  __shared__ _Float16 sV[2][32 * 16];   // shared, double-buffered (padded K rows)

  const int bh = blockIdx.x;
  const int b = bh >> 3, h = bh & 7;
  const int wave = threadIdx.x >> 5, lane = threadIdx.x & 31;
  const int half = lane >> 4, n = lane & 15;
  const int qrow0 = b * Tz + blockIdx.y * 128 + wave * 16;
  const int kbase_row = b * Tz;

  // zero everything once (pad regions are never rewritten)
  for (int i = lane; i < 16 * 32; i += 32) { sQ[wave][i] = (_Float16)0.f; sP[wave][i] = (_Float16)0.f; }
  for (int i = threadIdx.x; i < 2 * 16 * 32; i += 256) ((_Float16*)sK)[i] = (_Float16)0.f;
  for (int i = threadIdx.x; i < 2 * 32 * 16; i += 256) ((_Float16*)sV)[i] = (_Float16)0.f;

  // per-wave Q tile: 16 rows x 8 dims
  {
    const int r = lane & 15, c = (lane >> 4) * 4;
    const _Float16* src = qg + (size_t)(qrow0 + r) * Ez + h * DKz + c;
#pragma unroll
    for (int i = 0; i < 4; ++i) sQ[wave][r * 32 + c + i] = src[i];
  }
  __syncthreads();  // pads + Q visible; safe to start filling K/V tile 0

  // producer: wave 0, lanes 0-15 fetch K rows, lanes 16-31 fetch V rows.
  // One 16-byte (8-half) row per lane per tile.
#ifdef HAVE_ASYNC_LDS
#define ISSUE_TILE(kt, buf)                                                          \
  do {                                                                               \
    const int krow_ = kbase_row + (kt) * 16 + (lane & 15);                           \
    const _Float16* gsrc_ = ((lane < 16) ? kg : vg) + (size_t)krow_ * Ez + h * DKz;  \
    _Float16* ldst_ = (lane < 16) ? &sK[buf][(lane & 15) * 32]                       \
                                  : &sV[buf][(lane & 15) * 16];                      \
    async_copy16(gsrc_, ldst_);                                                      \
  } while (0)
#define WAIT_ASYNC() asm volatile("s_wait_asynccnt 0" ::: "memory")
#else
#define ISSUE_TILE(kt, buf)                                                          \
  do {                                                                               \
    const int krow_ = kbase_row + (kt) * 16 + (lane & 15);                           \
    const _Float16* gsrc_ = ((lane < 16) ? kg : vg) + (size_t)krow_ * Ez + h * DKz;  \
    _Float16* ldst_ = (lane < 16) ? &sK[buf][(lane & 15) * 32]                       \
                                  : &sV[buf][(lane & 15) * 16];                      \
    _Pragma("unroll") for (int i_ = 0; i_ < 8; ++i_) ldst_[i_] = gsrc_[i_];          \
    __builtin_prefetch(gsrc_ + 16 * Ez, 0, 3);                                       \
  } while (0)
#define WAIT_ASYNC() do { } while (0)
#endif

  if (wave == 0) ISSUE_TILE(0, 0);

  const v16h Aq = load_A(sQ[wave], 32, 0);
  float rmax[8], rsum[8];
#pragma unroll
  for (int r = 0; r < 8; ++r) { rmax[r] = -1e30f; rsum[r] = 0.f; }
  v8f acc = {};
  const float scale = 0.35355339059327373f;  // 1/sqrt(DK)

  if (wave == 0) WAIT_ASYNC();
  __syncthreads();  // tile 0 ready

  for (int kt = 0; kt < Tz / 16; ++kt) {
    const int cur = kt & 1;
    if (wave == 0 && kt + 1 < Tz / 16) ISSUE_TILE(kt + 1, cur ^ 1);

    v16h Bk = load_B_row(sK[cur], 32, 0, 0);
    v8f zero = {};
    v8f s = wmma_f16(Aq, Bk, zero);

#pragma unroll
    for (int r8 = 0; r8 < 8; ++r8) {
      float sv = s[r8] * scale;
      float bm = sv;
      bm = fmaxf(bm, __shfl_xor(bm, 1, 32));
      bm = fmaxf(bm, __shfl_xor(bm, 2, 32));
      bm = fmaxf(bm, __shfl_xor(bm, 4, 32));
      bm = fmaxf(bm, __shfl_xor(bm, 8, 32));
      const float nm = fmaxf(rmax[r8], bm);
      const float f = __expf(rmax[r8] - nm);
      const float p = __expf(sv - nm);
      float bs = p;
      bs += __shfl_xor(bs, 1, 32);
      bs += __shfl_xor(bs, 2, 32);
      bs += __shfl_xor(bs, 4, 32);
      bs += __shfl_xor(bs, 8, 32);
      rsum[r8] = rsum[r8] * f + bs;
      rmax[r8] = nm;
      acc[r8] *= f;
      sP[wave][(r8 + half * 8) * 32 + n] = (_Float16)p;  // C-layout -> A-layout via LDS
    }
    __builtin_amdgcn_wave_barrier();

    v16h Ap = load_A(sP[wave], 32, 0);
    v16h Bv = load_B_col(sV[cur], 16);
    acc = wmma_f16(Ap, Bv, acc);

    if (wave == 0) WAIT_ASYNC();   // next tile landed in LDS
    __syncthreads();               // publish next buffer / retire current
  }

  if (n < DKz) {
#pragma unroll
    for (int r = 0; r < 8; ++r) {
      const int m = r + half * 8;
      og[(size_t)(qrow0 + m) * Ez + h * DKz + n] = (_Float16)(acc[r] / rsum[r]);
    }
  }
#undef ISSUE_TILE
#undef WAIT_ASYNC
}

// ---------------------------------------------------------------------------
// Kernel 3: x1 = LN(x + attn @ Wo^T + bo); qout = cos(x1)*cos(theta_ffn)
// block = 256 (8 waves), 64 rows per block
// ---------------------------------------------------------------------------
__global__ void __launch_bounds__(256)
proj_ln_kernel(const _Float16* __restrict__ ag, const float* __restrict__ x,
               const float* __restrict__ Wo, const float* __restrict__ bo,
               const float* __restrict__ g1, const float* __restrict__ be1,
               const float* __restrict__ theta_ffn,
               float* __restrict__ x1g, _Float16* __restrict__ qog) {
  __shared__ _Float16 sW[Ez * Ez];
  __shared__ _Float16 sA[64 * Ez];
  __shared__ float    sO[64 * Ez];
  __shared__ float    sb[Ez];
  const int tid = threadIdx.x;
  const int row0 = blockIdx.x * 64;

  for (int i = tid; i < Ez * Ez; i += 256) { sW[i] = (_Float16)Wo[i]; sA[i] = ag[(size_t)row0 * Ez + i]; }
  if (tid < Ez) sb[tid] = bo[tid];
  __syncthreads();

  const int wave = tid >> 5, lane = tid & 31;
  const int half = lane >> 4, n = lane & 15;
  for (int job = wave; job < 16; job += 8) {
    const int rt = job >> 2, nbase = (job & 3) * 16;
    v8f acc = {};
#pragma unroll
    for (int kc = 0; kc < 2; ++kc) {
      v16h a = load_A(sA + rt * 16 * Ez, Ez, kc * 32);
      v16h b = load_B_row(sW, Ez, nbase, kc * 32);
      acc = wmma_f16(a, b, acc);
    }
#pragma unroll
    for (int r = 0; r < 8; ++r)
      sO[(rt * 16 + r + half * 8) * Ez + nbase + n] = acc[r] + sb[nbase + n];
  }
  __syncthreads();

  if (tid < 64) {
    const int grow = row0 + tid;
    const float cth = __cosf(*theta_ffn);
    float s1 = 0.f, s2 = 0.f;
    for (int j = 0; j < Ez; ++j) {
      const float v = sO[tid * Ez + j] + x[(size_t)grow * Ez + j];
      s1 += v; s2 += v * v;
    }
    const float mean = s1 * (1.f / Ez);
    const float var = s2 * (1.f / Ez) - mean * mean;
    const float inv = rsqrtf(var + 1e-5f);
    for (int j = 0; j < Ez; ++j) {
      const float v = sO[tid * Ez + j] + x[(size_t)grow * Ez + j];
      const float y = (v - mean) * inv * g1[j] + be1[j];
      x1g[(size_t)grow * Ez + j] = y;
      qog[(size_t)grow * Ez + j] = (_Float16)(__cosf(y) * cth);
    }
  }
}

// ---------------------------------------------------------------------------
// Kernel 4: out = LN(x1 + relu(qout @ W1^T + b1) @ W2^T + b2)
// block = 256 (8 waves), 32 rows per block; weights staged in shared (reused)
// ---------------------------------------------------------------------------
__global__ void __launch_bounds__(256)
ffn_kernel(const _Float16* __restrict__ qog, const float* __restrict__ x1g,
           const float* __restrict__ W1, const float* __restrict__ b1,
           const float* __restrict__ W2, const float* __restrict__ b2,
           const float* __restrict__ g2, const float* __restrict__ be2,
           float* __restrict__ out) {
  __shared__ _Float16 sWgt[FFz * Ez];   // 32 KB, W1 then W2
  __shared__ _Float16 sIn[32 * Ez];     // 4 KB
  __shared__ _Float16 sH[32 * FFz];     // 16 KB
  __shared__ float    sO[32 * Ez];      // 8 KB
  __shared__ float    sb1[FFz];
  __shared__ float    sb2[Ez];
  const int tid = threadIdx.x;
  const int row0 = blockIdx.x * 32;

  for (int i = tid; i < FFz * Ez; i += 256) sWgt[i] = (_Float16)W1[i];
  for (int i = tid; i < 32 * Ez; i += 256) sIn[i] = qog[(size_t)row0 * Ez + i];
  sb1[tid] = b1[tid];
  __syncthreads();

  const int wave = tid >> 5, lane = tid & 31;
  const int half = lane >> 4, n = lane & 15;

  // h = relu(qout @ W1^T + b1): 2 rowtiles x 16 ntiles
  for (int job = wave; job < 32; job += 8) {
    const int rt = job >> 4, nbase = (job & 15) * 16;
    v8f acc = {};
#pragma unroll
    for (int kc = 0; kc < 2; ++kc) {
      v16h a = load_A(sIn + rt * 16 * Ez, Ez, kc * 32);
      v16h b = load_B_row(sWgt, Ez, nbase, kc * 32);
      acc = wmma_f16(a, b, acc);
    }
#pragma unroll
    for (int r = 0; r < 8; ++r) {
      const int m = rt * 16 + r + half * 8;
      sH[m * FFz + nbase + n] = (_Float16)fmaxf(acc[r] + sb1[nbase + n], 0.f);
    }
  }
  __syncthreads();

  for (int i = tid; i < Ez * FFz; i += 256) sWgt[i] = (_Float16)W2[i];
  if (tid < Ez) sb2[tid] = b2[tid];
  __syncthreads();

  // f = h @ W2^T + b2: 2 rowtiles x 4 ntiles = 8 jobs, K = 256 (8 chunks)
  for (int job = wave; job < 8; job += 8) {
    const int rt = job >> 2, nbase = (job & 3) * 16;
    v8f acc = {};
#pragma unroll
    for (int kc = 0; kc < 8; ++kc) {
      v16h a = load_A(sH + rt * 16 * FFz, FFz, kc * 32);
      v16h b = load_B_row(sWgt, FFz, nbase, kc * 32);
      acc = wmma_f16(a, b, acc);
    }
#pragma unroll
    for (int r = 0; r < 8; ++r)
      sO[(rt * 16 + r + half * 8) * Ez + nbase + n] = acc[r] + sb2[nbase + n];
  }
  __syncthreads();

  if (tid < 32) {
    const int grow = row0 + tid;
    float s1 = 0.f, s2 = 0.f;
    for (int j = 0; j < Ez; ++j) {
      const float v = sO[tid * Ez + j] + x1g[(size_t)grow * Ez + j];
      s1 += v; s2 += v * v;
    }
    const float mean = s1 * (1.f / Ez);
    const float var = s2 * (1.f / Ez) - mean * mean;
    const float inv = rsqrtf(var + 1e-5f);
    for (int j = 0; j < Ez; ++j) {
      const float v = sO[tid * Ez + j] + x1g[(size_t)grow * Ez + j];
      out[(size_t)grow * Ez + j] = (v - mean) * inv * g2[j] + be2[j];
    }
  }
}

// ---------------------------------------------------------------------------
extern "C" void kernel_launch(void* const* d_in, const int* in_sizes, int n_in,
                              void* d_out, int out_size, void* d_ws, size_t ws_size,
                              hipStream_t stream) {
  const float* x   = (const float*)d_in[0];
  const float* Wq  = (const float*)d_in[1];
  const float* bq  = (const float*)d_in[2];
  const float* Wk  = (const float*)d_in[3];
  const float* bk  = (const float*)d_in[4];
  const float* Wv  = (const float*)d_in[5];
  const float* bv  = (const float*)d_in[6];
  const float* Wo  = (const float*)d_in[7];
  const float* bo  = (const float*)d_in[8];
  const float* tha = (const float*)d_in[9];
  const float* thf = (const float*)d_in[10];
  const float* W1  = (const float*)d_in[11];
  const float* b1  = (const float*)d_in[12];
  const float* W2  = (const float*)d_in[13];
  const float* b2  = (const float*)d_in[14];
  const float* g1  = (const float*)d_in[15];
  const float* be1 = (const float*)d_in[16];
  const float* g2  = (const float*)d_in[17];
  const float* be2 = (const float*)d_in[18];
  float* out = (float*)d_out;

  _Float16* qbuf  = (_Float16*)d_ws;
  _Float16* kbuf  = qbuf + (size_t)BT * Ez;
  _Float16* vbuf  = kbuf + (size_t)BT * Ez;
  _Float16* abuf  = vbuf + (size_t)BT * Ez;
  _Float16* qobuf = abuf + (size_t)BT * Ez;
  float*    x1buf = (float*)(qobuf + (size_t)BT * Ez);

  qkv_kernel<<<BT / 64, 256, 0, stream>>>(x, Wq, bq, Wk, bk, Wv, bv, tha,
                                          qbuf, kbuf, vbuf);
  attn_kernel<<<dim3(Bz * Hz, Tz / 128), 256, 0, stream>>>(qbuf, kbuf, vbuf, abuf);
  proj_ln_kernel<<<BT / 64, 256, 0, stream>>>(abuf, x, Wo, bo, g1, be1, thf,
                                              x1buf, qobuf);
  ffn_kernel<<<BT / 32, 256, 0, stream>>>(qobuf, x1buf, W1, b1, W2, b2, g2, be2,
                                          out);
}
